// VllmMixtureOfExpertsOp_65532611002731
// MI455X (gfx1250) — compile-verified, matrix operand order checked by reading, not run on hardware
//
#include <hip/hip_runtime.h>
#include <hip/hip_bf16.h>

// ---------------------------------------------------------------------------
// MoE FFN (SwiGLU) for MI455X / gfx1250, wave32 + v_wmma_f32_16x16x32_bf16.
// Compute-bound (~830 FLOP/byte) -> bf16 WMMA path, fp32 accumulation.
// v3: register double-buffered global->LDS + grouped B-fragment preload so
//     WMMAs issue back-to-back instead of stalling on per-fragment dscnt.
// ---------------------------------------------------------------------------

typedef __attribute__((ext_vector_type(16))) __bf16 v16bf;
typedef __attribute__((ext_vector_type(8)))  __bf16 v8bf;
typedef __attribute__((ext_vector_type(8)))  float  v8f;
typedef __attribute__((ext_vector_type(4)))  float  v4f;   // native clang vector
typedef unsigned short u16;
typedef unsigned int   u32;

static constexpr int T_TOK = 8192;
static constexpr int K_TOP = 2;
static constexpr int HD    = 1024;   // hidden
static constexpr int ID    = 2816;   // intermediate
static constexpr int NE    = 8;      // experts
static constexpr int N1    = 2 * ID; // 5632 (gate|up)
static constexpr int BM    = 128;    // M tile (token-entry rows)
static constexpr int PADT  = T_TOK * K_TOP + NE * BM;        // 17408 padded entries
static constexpr int MT_MAX = (T_TOK * K_TOP) / BM + NE;     // 136 m-tiles max
static constexpr int LDA   = 72;     // LDS strides (bf16 elems): 144B = 9-bank skew,
static constexpr int LDB   = 72;     // still 16B-aligned for b128 LDS accesses

__device__ __forceinline__ u16 f2bf(float f) {
    u32 u = __float_as_uint(f);
    u += 0x7FFFu + ((u >> 16) & 1u);          // round-to-nearest-even
    return (u16)(u >> 16);
}

// Assemble a 16x32 bf16 A (or [n][k]-staged B) fragment per ISA 7.12.2:
// lane<16 holds K {kb..kb+7, kb+16..kb+23}; lane>=16 holds {kb+8.., kb+24..}.
// With K contiguous in LDS these are two 16-byte ds loads per lane.
__device__ __forceinline__ v16bf load_frag(const u16* lds, int row, int ld,
                                           int kbase, int lane) {
    const int koff = kbase + ((lane & 16) ? 8 : 0);
    const __bf16* p = reinterpret_cast<const __bf16*>(lds) + row * ld + koff;
    v8bf lo = *reinterpret_cast<const v8bf*>(p);
    v8bf hi = *reinterpret_cast<const v8bf*>(p + 16);
    return __builtin_shufflevector(lo, hi, 0,1,2,3,4,5,6,7,8,9,10,11,12,13,14,15);
}

__device__ __forceinline__ v8f wmma_bf16(v16bf a, v16bf b, v8f c) {
    return __builtin_amdgcn_wmma_f32_16x16x32_bf16(false, a, false, b,
                                                   (short)0, c, false, false);
}

// ---------------------------------------------------------------------------
// fp32 -> bf16 bulk convert (weights read once: non-temporal loads)
// ---------------------------------------------------------------------------
__global__ void moe_convert_kernel(const float* __restrict__ src,
                                   u16* __restrict__ dst, size_t n4) {
    size_t i = (size_t)blockIdx.x * blockDim.x + threadIdx.x;
    const size_t stride = (size_t)gridDim.x * blockDim.x;
    for (; i < n4; i += stride) {
        v4f v = __builtin_nontemporal_load(reinterpret_cast<const v4f*>(src) + i);
        ushort4 o;
        o.x = f2bf(v.x); o.y = f2bf(v.y); o.z = f2bf(v.z); o.w = f2bf(v.w);
        *(reinterpret_cast<ushort4*>(dst) + i) = o;
    }
}

__global__ void moe_zero_kernel(float* __restrict__ p, size_t n4) {
    size_t i = (size_t)blockIdx.x * blockDim.x + threadIdx.x;
    const size_t stride = (size_t)gridDim.x * blockDim.x;
    const v4f z = {0.f, 0.f, 0.f, 0.f};
    for (; i < n4; i += stride) reinterpret_cast<v4f*>(p)[i] = z;
}

// ---------------------------------------------------------------------------
// Deterministic per-expert token-list build (counting sort, fixed scan order).
// Each expert segment is padded to a multiple of BM with token = -1.
// ---------------------------------------------------------------------------
__global__ __launch_bounds__(256)
void moe_route_kernel(const int* __restrict__ rt, const float* __restrict__ rw,
                      int* __restrict__ ltok, float* __restrict__ lcoef,
                      int* __restrict__ tile_off) {
    __shared__ int s_cnt[256][NE];
    __shared__ int s_base[256][NE];
    const int tid = threadIdx.x;
    const int per = (T_TOK * K_TOP) / 256;   // 64 entries / thread

    int lc[NE];
#pragma unroll
    for (int e = 0; e < NE; ++e) lc[e] = 0;
    for (int i = 0; i < per; ++i) lc[rt[tid * per + i]]++;
#pragma unroll
    for (int e = 0; e < NE; ++e) s_cnt[tid][e] = lc[e];
    __syncthreads();

    if (tid == 0) {
        int cnt[NE];
        for (int e = 0; e < NE; ++e) cnt[e] = 0;
        for (int t = 0; t < 256; ++t)
            for (int e = 0; e < NE; ++e) cnt[e] += s_cnt[t][e];
        int to[NE + 1]; to[0] = 0;
        for (int e = 0; e < NE; ++e) to[e + 1] = to[e] + (cnt[e] + BM - 1) / BM;
        for (int e = 0; e <= NE; ++e) tile_off[e] = to[e];
        int run[NE];
        for (int e = 0; e < NE; ++e) run[e] = to[e] * BM;     // padded seg start
        for (int t = 0; t < 256; ++t)
            for (int e = 0; e < NE; ++e) { s_base[t][e] = run[e]; run[e] += s_cnt[t][e]; }
    }
    __syncthreads();

    // init everything to padding, then overwrite real entries
    for (int i = tid; i < PADT; i += 256) { ltok[i] = -1; lcoef[i] = 0.f; }
    __syncthreads();

#pragma unroll
    for (int e = 0; e < NE; ++e) lc[e] = 0;
    for (int i = 0; i < per; ++i) {
        const int idx = tid * per + i;
        const int e = rt[idx];
        const int pos = s_base[tid][e] + lc[e]++;
        ltok[pos] = idx / K_TOP;         // token id
        lcoef[pos] = rw[idx];            // router weight
    }
}

// Shared WMMA tile-compute: A-fragment once, B-fragments preloaded in groups
// of 4 so DS loads pipeline ahead of bursts of back-to-back WMMAs.
__device__ __forceinline__ void wmma_tile(const u16* sA, const u16* sB,
                                          int w, int lane, v8f (&acc)[8]) {
#pragma unroll
    for (int kk = 0; kk < 64; kk += 32) {
        const v16bf af = load_frag(sA, w * 16 + (lane & 15), LDA, kk, lane);
#pragma unroll
        for (int jg = 0; jg < 2; ++jg) {
            const int jb = jg * 4;
            v16bf b0 = load_frag(sB, (jb + 0) * 16 + (lane & 15), LDB, kk, lane);
            v16bf b1 = load_frag(sB, (jb + 1) * 16 + (lane & 15), LDB, kk, lane);
            v16bf b2 = load_frag(sB, (jb + 2) * 16 + (lane & 15), LDB, kk, lane);
            v16bf b3 = load_frag(sB, (jb + 3) * 16 + (lane & 15), LDB, kk, lane);
            acc[jb + 0] = wmma_bf16(af, b0, acc[jb + 0]);
            acc[jb + 1] = wmma_bf16(af, b1, acc[jb + 1]);
            acc[jb + 2] = wmma_bf16(af, b2, acc[jb + 2]);
            acc[jb + 3] = wmma_bf16(af, b3, acc[jb + 3]);
        }
    }
}

// ---------------------------------------------------------------------------
// GEMM1: gu = gather(x) @ w13[e] for 64 gate cols + 64 up cols per block,
//        fused h = silu(gate)*up, bf16 h written to ws.
// Block = 256 threads = 8 waves. Wave w owns rows [16w,16w+16) x 128 cols.
// Register double-buffering: next K-tile global loads issue before compute.
// ---------------------------------------------------------------------------
__global__ __launch_bounds__(256)
void moe_gemm1_kernel(const u16* __restrict__ hsb, const u16* __restrict__ w13b,
                      const int* __restrict__ ltok, const int* __restrict__ toff,
                      u16* __restrict__ hb) {
    __shared__ u16 sA[BM * LDA];
    __shared__ u16 sB[BM * LDB];   // staged transposed: sB[n][k]
    __shared__ int sTok[BM];

    const int mt = blockIdx.y;
    if (mt >= toff[NE]) return;                 // block-uniform exit (EXEC stays full)
    int e = 0;
#pragma unroll
    for (int i = 0; i < NE - 1; ++i) if (mt >= toff[i + 1]) e = i + 1;

    const int n0 = blockIdx.x * 64;             // gate column base
    const int row_base = mt * BM;
    const int tid = threadIdx.x;
    const int lane = tid & 31;
    const int w = tid >> 5;

    if (tid < BM) sTok[tid] = ltok[row_base + tid];
    __syncthreads();

    v8f acc[8];
    const v8f vz = {0.f,0.f,0.f,0.f,0.f,0.f,0.f,0.f};
#pragma unroll
    for (int j = 0; j < 8; ++j) acc[j] = vz;

    const int arow  = tid >> 1;                 // 2 threads per A row
    const int ahalf = (tid & 1) * 32;
    const int tokv  = sTok[arow];
    const size_t abase = (size_t)(tokv < 0 ? 0 : tokv) * HD;  // pad rows read row 0

    const int bk = tid >> 2;                    // 4 threads per B k-row
    const int bq = (tid & 3) * 32;              // combined-column quarter

    uint4 ra[4], rb[4];
    auto load_tiles = [&](int k0) {
#pragma unroll
        for (int u = 0; u < 4; ++u)
            ra[u] = *reinterpret_cast<const uint4*>(hsb + abase + k0 + ahalf + u * 8);
        const size_t rbase = ((size_t)e * HD + (size_t)(k0 + bk)) * (size_t)N1;
#pragma unroll
        for (int u = 0; u < 4; ++u) {
            const int c0 = bq + u * 8;
            const int gcol = (c0 < 64) ? (n0 + c0) : (ID + n0 + (c0 - 64));
            rb[u] = *reinterpret_cast<const uint4*>(w13b + rbase + gcol);
        }
        if (k0 + 64 < HD) {                     // warm L2 one tile further ahead
            __builtin_prefetch(hsb + abase + k0 + 64 + ahalf, 0, 1);
            __builtin_prefetch(w13b + rbase + 64 * (size_t)N1 + n0 + bq, 0, 1);
        }
    };

    load_tiles(0);
    for (int k0 = 0; k0 < HD; k0 += 64) {
        __syncthreads();                        // previous tile's compute done
#pragma unroll
        for (int u = 0; u < 4; ++u)             // staged regs -> LDS (b128)
            *reinterpret_cast<uint4*>(&sA[arow * LDA + ahalf + u * 8]) = ra[u];
#pragma unroll
        for (int u = 0; u < 4; ++u) {           // B transposed into sB[n][k]
            const int c0 = bq + u * 8;
            union { uint4 v; u16 s[8]; } x;
            x.v = rb[u];
#pragma unroll
            for (int jj = 0; jj < 8; ++jj) sB[(c0 + jj) * LDB + bk] = x.s[jj];
        }
        __syncthreads();                        // LDS tile ready
        if (k0 + 64 < HD) load_tiles(k0 + 64);  // overlap next global loads w/ WMMA
        wmma_tile(sA, sB, w, lane, acc);
    }

    // Epilogue: tiles j=0..3 are gate, j+4 the matching up. C layout: lane->N, vgpr->M.
    const int nlane = lane & 15;
    const int rofs  = (lane >> 4) * 8;
#pragma unroll
    for (int j = 0; j < 4; ++j) {
#pragma unroll
        for (int r = 0; r < 8; ++r) {
            const float g = acc[j][r];
            const float u = acc[j + 4][r];
            const float h = (g / (1.f + __expf(-g))) * u;     // silu(g)*u
            const int pos = row_base + w * 16 + r + rofs;
            const int col = n0 + j * 16 + nlane;
            hb[(size_t)pos * ID + col] = f2bf(h);
        }
    }
}

// ---------------------------------------------------------------------------
// GEMM2: out[token] += coeff * (h @ w2[e]), 128x128 tile, atomic f32 adds
// (exactly 2 commutative adds per output element -> deterministic).
// ---------------------------------------------------------------------------
__global__ __launch_bounds__(256)
void moe_gemm2_kernel(const u16* __restrict__ hb, const u16* __restrict__ w2b,
                      const int* __restrict__ ltok, const float* __restrict__ lcoef,
                      const int* __restrict__ toff, float* __restrict__ out) {
    __shared__ u16 sA[BM * LDA];
    __shared__ u16 sB[BM * LDB];
    __shared__ int   sTok[BM];
    __shared__ float sCf[BM];

    const int mt = blockIdx.y;
    if (mt >= toff[NE]) return;
    int e = 0;
#pragma unroll
    for (int i = 0; i < NE - 1; ++i) if (mt >= toff[i + 1]) e = i + 1;

    const int n0 = blockIdx.x * 128;
    const int row_base = mt * BM;
    const int tid = threadIdx.x;
    const int lane = tid & 31;
    const int w = tid >> 5;

    if (tid < BM) { sTok[tid] = ltok[row_base + tid]; sCf[tid] = lcoef[row_base + tid]; }
    __syncthreads();

    v8f acc[8];
    const v8f vz = {0.f,0.f,0.f,0.f,0.f,0.f,0.f,0.f};
#pragma unroll
    for (int j = 0; j < 8; ++j) acc[j] = vz;

    const int arow  = tid >> 1;
    const int ahalf = (tid & 1) * 32;
    const size_t abase = (size_t)(row_base + arow) * ID;   // h rows are contiguous
    const int bk = tid >> 2;
    const int bq = (tid & 3) * 32;

    uint4 ra[4], rb[4];
    auto load_tiles = [&](int k0) {
#pragma unroll
        for (int u = 0; u < 4; ++u)
            ra[u] = *reinterpret_cast<const uint4*>(hb + abase + k0 + ahalf + u * 8);
        const size_t rbase = ((size_t)e * ID + (size_t)(k0 + bk)) * (size_t)HD;
#pragma unroll
        for (int u = 0; u < 4; ++u)
            rb[u] = *reinterpret_cast<const uint4*>(w2b + rbase + n0 + bq + u * 8);
        if (k0 + 64 < ID) {
            __builtin_prefetch(hb + abase + k0 + 64 + ahalf, 0, 1);
            __builtin_prefetch(w2b + rbase + 64 * (size_t)HD + n0 + bq, 0, 1);
        }
    };

    load_tiles(0);
    for (int k0 = 0; k0 < ID; k0 += 64) {
        __syncthreads();
#pragma unroll
        for (int u = 0; u < 4; ++u)
            *reinterpret_cast<uint4*>(&sA[arow * LDA + ahalf + u * 8]) = ra[u];
#pragma unroll
        for (int u = 0; u < 4; ++u) {
            const int c0 = bq + u * 8;
            union { uint4 v; u16 s[8]; } x;
            x.v = rb[u];
#pragma unroll
            for (int jj = 0; jj < 8; ++jj) sB[(c0 + jj) * LDB + bk] = x.s[jj];
        }
        __syncthreads();
        if (k0 + 64 < ID) load_tiles(k0 + 64);
        wmma_tile(sA, sB, w, lane, acc);
    }

    const int nlane = lane & 15;
    const int rofs  = (lane >> 4) * 8;
#pragma unroll
    for (int j = 0; j < 8; ++j) {
#pragma unroll
        for (int r = 0; r < 8; ++r) {
            const int m = w * 16 + r + rofs;
            const int tokv = sTok[m];
            if (tokv < 0) continue;                           // padding row
            const int col = n0 + j * 16 + nlane;
            unsafeAtomicAdd(&out[(size_t)tokv * HD + col], sCf[m] * acc[j][r]);
        }
    }
}

// ---------------------------------------------------------------------------
// Host launcher
// ---------------------------------------------------------------------------
extern "C" void kernel_launch(void* const* d_in, const int* in_sizes, int n_in,
                              void* d_out, int out_size, void* d_ws, size_t ws_size,
                              hipStream_t stream) {
    const float* hs  = (const float*)d_in[0];
    const int*   rt  = (const int*)d_in[1];
    const float* rw  = (const float*)d_in[2];
    const float* w13 = (const float*)d_in[3];
    const float* w2  = (const float*)d_in[4];
    float* out = (float*)d_out;

    const size_t HSB_N = (size_t)T_TOK * HD;          //  8.4 M bf16
    const size_t W13_N = (size_t)NE * HD * N1;        // 46.1 M bf16
    const size_t W2_N  = (size_t)NE * ID * HD;        // 23.1 M bf16
    const size_t HB_N  = (size_t)PADT * ID;           // 49.0 M bf16

    size_t off = 0;
    auto carve = [&](size_t bytes) { size_t o = off; off = (off + bytes + 255) & ~(size_t)255; return o; };
    char* ws = (char*)d_ws;
    u16*   hsb   = (u16*)  (ws + carve(HSB_N * 2));
    u16*   w13b  = (u16*)  (ws + carve(W13_N * 2));
    u16*   w2b   = (u16*)  (ws + carve(W2_N * 2));
    u16*   hb    = (u16*)  (ws + carve(HB_N * 2));
    int*   ltok  = (int*)  (ws + carve(PADT * sizeof(int)));
    float* lcoef = (float*)(ws + carve(PADT * sizeof(float)));
    int*   toff  = (int*)  (ws + carve((NE + 1) * sizeof(int)));
    if (off > ws_size) return;   // needs ~254 MB of scratch

    // 1) bf16 conversions (weights resident in 192MB L2 afterwards)
    moe_convert_kernel<<<2048, 256, 0, stream>>>(hs,  hsb,  HSB_N / 4);
    moe_convert_kernel<<<4096, 256, 0, stream>>>(w13, w13b, W13_N / 4);
    moe_convert_kernel<<<4096, 256, 0, stream>>>(w2,  w2b,  W2_N  / 4);
    // 2) deterministic routing sort
    moe_route_kernel<<<1, 256, 0, stream>>>(rt, rw, ltok, lcoef, toff);
    // 3) zero output (it is poisoned; gemm2 accumulates atomically)
    moe_zero_kernel<<<2048, 256, 0, stream>>>(out, (size_t)T_TOK * HD / 4);
    // 4) x @ w13 + SwiGLU
    moe_gemm1_kernel<<<dim3(ID / 64, MT_MAX), 256, 0, stream>>>(hsb, w13b, ltok, toff, hb);
    // 5) h @ w2, scaled accumulate into out
    moe_gemm2_kernel<<<dim3(HD / 128, MT_MAX), 256, 0, stream>>>(hb, w2b, ltok, lcoef, toff, out);
}
